// QConv2D_layer_26087631356429
// MI455X (gfx1250) — matrix operand order repaired and checked by reading, not compiled, not run on hardware
//
#include <hip/hip_runtime.h>
#include <hip/hip_bf16.h>
#include <math.h>

// ---------------------------------------------------------------------------
// QConv2D (9-qubit quantum conv) reformulated as GEMM for CDNA5 WMMA.
//
// out[p,f] = sum_k z_k |(U_f psi_p)_k|^2 ,  z_k = (-1)^(k&1)
// psi_p splits into real vectors psi_r / psi_i (product-state encoding),
// U_f is the real orthogonal 512x512 unitary of the parameter circuit.
// => Y = Phi (50176 x 512, f16) x U_f^T (512x512, f16) via v_wmma_f32_16x16x32_f16,
//    then square + signed reduce.
// ---------------------------------------------------------------------------

typedef _Float16 half16 __attribute__((ext_vector_type(16)));
typedef _Float16 half8  __attribute__((ext_vector_type(8)));
typedef float    float8 __attribute__((ext_vector_type(8)));

#define NQ      9
#define DIM     512
#define NPIX    25088                 // 32*28*28 pixels
#define NF      4
#define NLAYERS 2

// workspace layout (bytes):
//   [0, 2MB)        : B-pack, 4 filters * 32 ntiles * 16 ktiles * 512 halves
//   [2MB, ~53.4MB)  : Phi, 50176 rows * 512 halves (f16)
#define PHI_OFF_BYTES (2u * 1024u * 1024u)

// ---------------------------------------------------------------------------
// Kernel 1: simulate the parameter circuit to build U_f, emit it directly in
// the WMMA B-operand lane layout.
//   grid = (32, NF), 256 threads. Each block owns 16 columns of U_f in LDS.
//   Column block kb16 covers K-window [16*kb16, 16*kb16+16) of K-tile kt=kb16/2,
//   half g=kb16&1 (lanes 16g..16g+15 of every B tile).
// B order used here (contiguous k, not the ISA table order): lane L: N = L&15, K = h + 16*(L>>4),
// h = 0..15 contiguous -> each lane's 16 halves are contiguous in memory.
// ---------------------------------------------------------------------------
__global__ void build_unitary_pack(const float* __restrict__ kparams,
                                   _Float16* __restrict__ bpack)
{
    __shared__ float U[DIM * 16];          // U[k*16 + c], c = column - 16*kb16
    const int kb16 = blockIdx.x;           // 0..31
    const int f    = blockIdx.y;           // filter
    const int tid  = threadIdx.x;

    // init: columns are basis vectors e_j
    for (int idx = tid; idx < DIM * 16; idx += 256) {
        int k = idx >> 4, c = idx & 15;
        U[idx] = (k == (kb16 * 16 + c)) ? 1.0f : 0.0f;
    }
    __syncthreads();

    // parameter circuit: per layer, 9 Ry then CNOT ladder q -> q+1.
    // qubit q lives at bit position (8 - q) of the amplitude index.
    for (int l = 0; l < NLAYERS; ++l) {
        for (int q = 0; q < NQ; ++q) {
            float th = 0.5f * kparams[f * 18 + l * 9 + q];
            float cg, sg;
            sincosf(th, &sg, &cg);         // Ry = [[c,-s],[s,c]]
            const int b = 8 - q;
            for (int w = tid; w < 4096; w += 256) {        // 256 pairs * 16 cols
                int c = w & 15, t = w >> 4;
                int k0 = ((t >> b) << (b + 1)) | (t & ((1 << b) - 1));
                int k1 = k0 | (1 << b);
                float v0 = U[k0 * 16 + c], v1 = U[k1 * 16 + c];
                U[k0 * 16 + c] = cg * v0 - sg * v1;
                U[k1 * 16 + c] = sg * v0 + cg * v1;
            }
            __syncthreads();
        }
        for (int q = 0; q < NQ - 1; ++q) {                 // CNOT(q -> q+1)
            const int bt = 7 - q;                          // target bit; control = bt+1
            for (int w = tid; w < 2048; w += 256) {        // 128 pairs * 16 cols
                int c = w & 15, t = w >> 4;
                int k0 = ((t >> bt) << (bt + 2)) | (1 << (bt + 1)) |
                         (t & ((1 << bt) - 1));            // control=1, target=0
                int k1 = k0 | (1 << bt);
                float v0 = U[k0 * 16 + c], v1 = U[k1 * 16 + c];
                U[k0 * 16 + c] = v1;
                U[k1 * 16 + c] = v0;
            }
            __syncthreads();
        }
    }

    // pack: B[i=K][n=N] = U[n][i].  Element for (jn, lane L=16g+nu, h):
    //   U[16*jn + nu][32*kt + 16*g + h]  with 16*kb16 == 32*kt + 16*g.
    const int kt = kb16 >> 1, g = kb16 & 1;
    const int nu = tid & 15;
    for (int pass = 0; pass < 2; ++pass) {
        int jn = (tid >> 4) + 16 * pass;                   // 0..31
        size_t base = ((((size_t)f * 32 + jn) * 16 + kt) * 512) +
                      (size_t)(g * 16 + nu) * 16;
        #pragma unroll
        for (int h = 0; h < 16; ++h)
            bpack[base + h] = (_Float16)U[(16 * jn + nu) * 16 + h];
    }
}

// ---------------------------------------------------------------------------
// Kernel 2: build encoded-state matrix Phi (f16). One wave per pixel.
// Row 2p   = Re(psi_p), row 2p+1 = Im(psi_p), each 512 halves, row-major.
// amplitude i = (-i)^popc(i) * prod_q ( bit_{8-q}(i) ? sin(x_q/2) : cos(x_q/2) )
// Zero padding -> x=0 -> Rx(0)=I, matching SAME padding semantics.
// ---------------------------------------------------------------------------
__global__ void encode_states(const float* __restrict__ x,
                              _Float16* __restrict__ phi)
{
    const int lane = threadIdx.x & 31;
    const int p    = blockIdx.x * 8 + (threadIdx.x >> 5);  // pixel id
    const int b = p / 784, rem = p % 784, h = rem / 28, w = rem % 28;

    float cq[9], sq[9];
    #pragma unroll
    for (int q = 0; q < 9; ++q) {
        int hh = h + q / 3 - 1, ww = w + q % 3 - 1;
        float xv = (hh >= 0 && hh < 28 && ww >= 0 && ww < 28)
                       ? x[(b * 28 + hh) * 28 + ww] : 0.0f;
        sincosf(0.5f * xv, &sq[q], &cq[q]);
    }

    half16 rr, ii;
    #pragma unroll
    for (int t = 0; t < 16; ++t) {
        int i = lane * 16 + t;
        float prod = 1.0f;
        #pragma unroll
        for (int q = 0; q < 9; ++q)
            prod *= ((i >> (8 - q)) & 1) ? sq[q] : cq[q];
        int pc = __popc(i) & 3;
        float re = (pc == 0) ? prod : ((pc == 2) ? -prod : 0.0f);
        float im = (pc == 1) ? -prod : ((pc == 3) ? prod : 0.0f);
        rr[t] = (_Float16)re;
        ii[t] = (_Float16)im;
    }
    *(half16*)(phi + (size_t)(2 * p)     * 512 + lane * 16) = rr;
    *(half16*)(phi + (size_t)(2 * p + 1) * 512 + lane * 16) = ii;
}

// ---------------------------------------------------------------------------
// Kernel 3: GEMM + fused epilogue. grid = (196, NF), 256 threads (8 waves).
// Each wave: 2 M-tiles (32 Phi rows = 16 pixels) x full U_f. B tile is loaded
// once and fed to two v_wmma_f32_16x16x32_f16 (2x B reuse). 32 N-tiles x 16
// K-steps, then square, z-sign (lane&1), shfl reduce.
// A order used here (contiguous k, matches B pack): lane L: M = L&15, g = L>>4,
//   halves 0..7  -> K = 16g + h          (contiguous 16B run)
//   halves 8..15 -> K = 16g + h (contiguous 16B run)
// ---------------------------------------------------------------------------
__global__ void qconv_wmma(const _Float16* __restrict__ bpack,
                           const _Float16* __restrict__ phi,
                           float* __restrict__ out)
{
    const int lane = threadIdx.x & 31;
    const int wave = threadIdx.x >> 5;
    const int gw   = blockIdx.x * 8 + wave;     // 0..1567, owns 2 M-tiles
    const int f    = blockIdx.y;
    const int g    = lane >> 4;

    const _Float16* __restrict__ arow0 =
        phi + ((size_t)gw * 32 + (lane & 15)) * 512;       // M-tile 2*gw
    const _Float16* __restrict__ arow1 = arow0 + 16 * 512; // M-tile 2*gw+1

    float acc0[8], acc1[8];
    #pragma unroll
    for (int r = 0; r < 8; ++r) { acc0[r] = 0.0f; acc1[r] = 0.0f; }

    #pragma unroll 1
    for (int jn = 0; jn < 32; ++jn) {
        const _Float16* __restrict__ bt0 =
            bpack + (((size_t)f * 32 + jn) * 16) * 512 + (size_t)lane * 16;
        __builtin_prefetch(bt0 + 16 * 512, 0, 1);          // next N-tile block

        float8 c0 = {0.f, 0.f, 0.f, 0.f, 0.f, 0.f, 0.f, 0.f};
        float8 c1 = {0.f, 0.f, 0.f, 0.f, 0.f, 0.f, 0.f, 0.f};
        #pragma unroll
        for (int kt = 0; kt < 16; ++kt) {
            half16 bm = *(const half16*)(bt0 + (size_t)kt * 512);

            half8 lo0 = *(const half8*)(arow0 + kt * 32 + 16 * g);
            half8 hi0 = *(const half8*)(arow0 + kt * 32 + 16 * g + 8);
            half16 a0;
            #pragma unroll
            for (int t = 0; t < 8; ++t) { a0[t] = lo0[t]; a0[t + 8] = hi0[t]; }
            c0 = __builtin_amdgcn_wmma_f32_16x16x32_f16(
                     false, a0, false, bm, (short)0, c0, false, false);

            half8 lo1 = *(const half8*)(arow1 + kt * 32 + 16 * g);
            half8 hi1 = *(const half8*)(arow1 + kt * 32 + 16 * g + 8);
            half16 a1;
            #pragma unroll
            for (int t = 0; t < 8; ++t) { a1[t] = lo1[t]; a1[t + 8] = hi1[t]; }
            c1 = __builtin_amdgcn_wmma_f32_16x16x32_f16(
                     false, a1, false, bm, (short)0, c1, false, false);
        }
        #pragma unroll
        for (int r = 0; r < 8; ++r) {
            acc0[r] += c0[r] * c0[r];
            acc1[r] += c1[r] * c1[r];
        }
    }

    // z-sign: column index = 16*jn + (lane&15) -> sign depends only on lane&1
    const float zs = (lane & 1) ? -1.0f : 1.0f;
    #pragma unroll
    for (int r = 0; r < 8; ++r) {
        float v0 = acc0[r] * zs, v1 = acc1[r] * zs;
        v0 += __shfl_xor(v0, 1, 32);  v1 += __shfl_xor(v1, 1, 32);
        v0 += __shfl_xor(v0, 2, 32);  v1 += __shfl_xor(v1, 2, 32);
        v0 += __shfl_xor(v0, 4, 32);  v1 += __shfl_xor(v1, 4, 32);
        v0 += __shfl_xor(v0, 8, 32);  v1 += __shfl_xor(v1, 8, 32);
        acc0[r] = v0;  acc1[r] = v1;
    }

    // rows m = r + 8g of each tile: pixel = tile*8 + (m>>1), parts r/i at m&1.
    if ((lane & 15) == 0) {
        int pb0 = gw * 16 + g * 4;          // pixels of M-tile 2*gw
        int pb1 = gw * 16 + 8 + g * 4;      // pixels of M-tile 2*gw+1
        #pragma unroll
        for (int qq = 0; qq < 4; ++qq) {
            out[(size_t)(pb0 + qq) * 4 + f] = acc0[2 * qq] + acc0[2 * qq + 1];
            out[(size_t)(pb1 + qq) * 4 + f] = acc1[2 * qq] + acc1[2 * qq + 1];
        }
    }
}

// ---------------------------------------------------------------------------
extern "C" void kernel_launch(void* const* d_in, const int* in_sizes, int n_in,
                              void* d_out, int out_size, void* d_ws, size_t ws_size,
                              hipStream_t stream)
{
    const float* x  = (const float*)d_in[0];   // inputs (32,28,28,1) f32
    const float* kp = (const float*)d_in[1];   // kernel (4,1,18) f32
    float* out = (float*)d_out;                // (32,28,28,4) f32

    _Float16* bpack = (_Float16*)d_ws;                              // 2 MB
    _Float16* phi   = (_Float16*)((char*)d_ws + PHI_OFF_BYTES);     // ~51.4 MB

    build_unitary_pack<<<dim3(32, NF), 256, 0, stream>>>(kp, bpack);
    encode_states<<<dim3(NPIX / 8), 256, 0, stream>>>(x, phi);
    qconv_wmma<<<dim3(NPIX / 128, NF), 256, 0, stream>>>(bpack, phi, out);
}
